// DualModalExpertContainer_73890617360574
// MI455X (gfx1250) — matrix operand order repaired and checked by reading, not compiled, not run on hardware
//
#include <hip/hip_runtime.h>

typedef __attribute__((ext_vector_type(16))) _Float16 v16h;
typedef __attribute__((ext_vector_type(8)))  _Float16 v8h;
typedef __attribute__((ext_vector_type(8)))  float    v8f;

__device__ __forceinline__ float silu_f(float y) {
  return y / (1.0f + __expf(-y));
}

// One expert GEMM: A = f32 weights [256][Kdim] (row-major, converted to f16 on the
// fly), B = f16 activations in LDS, [pixel][K] with row stride BSTRIDE halves.
// Wave computes a 16(M) x 64(N) strip as 4 WMMA tiles, then applies BN+SiLU and
// accumulates ce * result into accOut.
template<int KSTEPS, int BSTRIDE>
__device__ __forceinline__ void gemm_expert(
    const float* __restrict__ Aw,
    const _Float16* Bb,
    const float* sInvE, const float* sBetaE,
    float ce, int Mbase, int l15, int hiK, v8f accOut[4])
{
  constexpr int Kdim = KSTEPS * 32;
  v8f zero8 = {0.f,0.f,0.f,0.f,0.f,0.f,0.f,0.f};
  v8f acc[4] = {zero8, zero8, zero8, zero8};

  // A fragment (16x32 f16): lanes 0-15 hold M=lane, K-chunks {0..7} and {16..23};
  // lanes 16-31 hold M=lane-16, K-chunks {8..15} and {24..31}.
  const float* arow = Aw + (size_t)(Mbase + l15) * Kdim + hiK * 8;
#pragma unroll
  for (int ks = 0; ks < KSTEPS; ++ks) {
    const float* ap = arow + ks * 32;
    float4 p0 = *(const float4*)(ap + 0);
    float4 p1 = *(const float4*)(ap + 4);
    float4 p2 = *(const float4*)(ap + 16);
    float4 p3 = *(const float4*)(ap + 20);
    v16h A;
    A[0]=(_Float16)p0.x;  A[1]=(_Float16)p0.y;  A[2]=(_Float16)p0.z;  A[3]=(_Float16)p0.w;
    A[4]=(_Float16)p1.x;  A[5]=(_Float16)p1.y;  A[6]=(_Float16)p1.z;  A[7]=(_Float16)p1.w;
    A[8]=(_Float16)p2.x;  A[9]=(_Float16)p2.y;  A[10]=(_Float16)p2.z; A[11]=(_Float16)p2.w;
    A[12]=(_Float16)p3.x; A[13]=(_Float16)p3.y; A[14]=(_Float16)p3.z; A[15]=(_Float16)p3.w;

#pragma unroll
    for (int nt = 0; nt < 4; ++nt) {
      // B fragment (32x16 f16): lane l<16 holds column N=l, K-chunks {0..7},{16..23};
      // lane>=16 holds column N=l-16, K-chunks {8..15},{24..31} (within this K-step).
      const _Float16* bp = Bb + (nt*16 + l15) * BSTRIDE + ks*32 + hiK*8;
      v8h blo = *(const v8h*)(bp);
      v8h bhi = *(const v8h*)(bp + 16);
      v16h Bf;
#pragma unroll
      for (int j = 0; j < 8; ++j) { Bf[j] = blo[j]; Bf[j+8] = bhi[j]; }
      acc[nt] = __builtin_amdgcn_wmma_f32_16x16x32_f16(
          false, A, false, Bf, (short)0, acc[nt], false, false);
    }
  }

  // C/D layout: VGPR v -> M = v + 8*(lane>=16), N = lane%16.
  const int obase = Mbase + hiK * 8;
#pragma unroll
  for (int v = 0; v < 8; ++v) {
    float inv = sInvE[obase + v];
    float bet = sBetaE[obase + v];
#pragma unroll
    for (int nt = 0; nt < 4; ++nt) {
      float y = acc[nt][v] * inv + bet;
      accOut[nt][v] += ce * silu_f(y);
    }
  }
}

__global__ __launch_bounds__(512)
void dual_modal_expert_kernel(
    const float* __restrict__ x,
    const float* __restrict__ weights,
    const int*   __restrict__ indices,
    const float* __restrict__ dw_w0,
    const float* __restrict__ pw_w0,
    const float* __restrict__ bn0_s, const float* __restrict__ bn0_b,
    const float* __restrict__ bn0_m, const float* __restrict__ bn0_v,
    const float* __restrict__ dw_w1,
    const float* __restrict__ pw_w1,
    const float* __restrict__ bn1_s, const float* __restrict__ bn1_b,
    const float* __restrict__ bn1_m, const float* __restrict__ bn1_v,
    const float* __restrict__ w2,
    const float* __restrict__ bn2_s, const float* __restrict__ bn2_b,
    const float* __restrict__ bn2_m, const float* __restrict__ bn2_v,
    float* __restrict__ out)
{
  // LDS (f16 act tiles, strides chosen so dword-stride % 64 == 4 -> conflict-free
  // b128 fragment reads). Total ~75 KB, fits easily in CDNA5's 320 KB/WGP.
  __shared__ __align__(16) _Float16 sB2[64 * 264];   // x row, [w][c], c=0..255
  __shared__ __align__(16) _Float16 sB0[64 * 136];   // dwconv half0, [w][k]
  __shared__ __align__(16) _Float16 sB1[64 * 136];   // dwconv half1, [w][k]
  __shared__ float sInv[3 * 256];
  __shared__ float sBeta[3 * 256];

  const int tid = threadIdx.x;
  const int bh  = blockIdx.x;
  const int b   = bh >> 6;
  const int h   = bh & 63;

  // routing coefficients (TOPK = 2, experts 0..2 used)
  float wa = weights[b*2 + 0], wb = weights[b*2 + 1];
  int   ia = indices[b*2 + 0], ib = indices[b*2 + 1];
  float coef0 = (ia == 0 ? wa : 0.f) + (ib == 0 ? wb : 0.f);
  float coef1 = (ia == 1 ? wa : 0.f) + (ib == 1 ? wb : 0.f);
  float coef2 = (ia == 2 ? wa : 0.f) + (ib == 2 ? wb : 0.f);

  // fold BN params: y*inv + beta
  for (int i = tid; i < 768; i += 512) {
    int e = i >> 8, o = i & 255;
    const float *ps, *pb, *pm, *pv;
    if (e == 0)      { ps = bn0_s; pb = bn0_b; pm = bn0_m; pv = bn0_v; }
    else if (e == 1) { ps = bn1_s; pb = bn1_b; pm = bn1_m; pv = bn1_v; }
    else             { ps = bn2_s; pb = bn2_b; pm = bn2_m; pv = bn2_v; }
    float inv = ps[o] * rsqrtf(pv[o] + 1e-5f);
    sInv[i]  = inv;
    sBeta[i] = pb[o] - pm[o] * inv;
  }

  // stage x[b,:,h,:] transposed to f16 [w][c]
  {
    const float* xb = x + ((size_t)b * 256 * 4096) + (size_t)h * 64;
    for (int i = tid; i < 256 * 64; i += 512) {
      int c = i >> 6, w = i & 63;
      sB2[w * 264 + c] = (_Float16)xb[(size_t)c * 4096 + w];
    }
  }

  // depthwise 3x3 (pad 1) on each half -> f16 [w][k]
  for (int i = tid; i < 2 * 128 * 64; i += 512) {
    int hs = i >> 13;
    int k  = (i >> 6) & 127;
    int w  = i & 63;
    const float* dwp = (hs ? dw_w1 : dw_w0) + k * 9;
    const float* xc  = x + (((size_t)b * 256 + hs * 128 + k) * 4096);
    float acc = 0.f;
#pragma unroll
    for (int dy = 0; dy < 3; ++dy) {
      int hh = h + dy - 1;
      if (hh < 0 || hh > 63) continue;
#pragma unroll
      for (int dx = 0; dx < 3; ++dx) {
        int ww = w + dx - 1;
        if (ww < 0 || ww > 63) continue;
        acc += dwp[dy*3 + dx] * xc[hh*64 + ww];
      }
    }
    if (hs) sB1[w * 136 + k] = (_Float16)acc;
    else    sB0[w * 136 + k] = (_Float16)acc;
  }

  __syncthreads();

  // phase 2: 16 waves, wave wv owns out-channels [16*wv, 16*wv+16) x 64 pixels
  const int wv    = tid >> 5;
  const int lane  = tid & 31;
  const int Mbase = wv * 16;
  const int l15   = lane & 15;
  const int hiK   = (lane >> 4) & 1;

  v8f zero8 = {0.f,0.f,0.f,0.f,0.f,0.f,0.f,0.f};
  v8f accOut[4] = {zero8, zero8, zero8, zero8};

  gemm_expert<4, 136>(pw_w0, sB0, sInv + 0,   sBeta + 0,   coef0, Mbase, l15, hiK, accOut);
  gemm_expert<4, 136>(pw_w1, sB1, sInv + 256, sBeta + 256, coef1, Mbase, l15, hiK, accOut);
  gemm_expert<8, 264>(w2,    sB2, sInv + 512, sBeta + 512, coef2, Mbase, l15, hiK, accOut);

  // store: out[b][o][h][w], lanes 0-15 cover consecutive w -> coalesced
  const int obase = Mbase + hiK * 8;
  float* op = out + (((size_t)b * 256 + obase) * 64 + h) * 64;
#pragma unroll
  for (int v = 0; v < 8; ++v) {
#pragma unroll
    for (int nt = 0; nt < 4; ++nt) {
      op[(size_t)v * 4096 + nt * 16 + l15] = accOut[nt][v];
    }
  }
}

extern "C" void kernel_launch(void* const* d_in, const int* in_sizes, int n_in,
                              void* d_out, int out_size, void* d_ws, size_t ws_size,
                              hipStream_t stream) {
  const float* x     = (const float*)d_in[0];
  const float* wts   = (const float*)d_in[1];
  const int*   idx   = (const int*)d_in[2];
  const float* dw_w0 = (const float*)d_in[3];
  const float* pw_w0 = (const float*)d_in[4];
  const float* bn0_s = (const float*)d_in[5];
  const float* bn0_b = (const float*)d_in[6];
  const float* bn0_m = (const float*)d_in[7];
  const float* bn0_v = (const float*)d_in[8];
  const float* dw_w1 = (const float*)d_in[9];
  const float* pw_w1 = (const float*)d_in[10];
  const float* bn1_s = (const float*)d_in[11];
  const float* bn1_b = (const float*)d_in[12];
  const float* bn1_m = (const float*)d_in[13];
  const float* bn1_v = (const float*)d_in[14];
  const float* w2    = (const float*)d_in[15];
  const float* bn2_s = (const float*)d_in[16];
  const float* bn2_b = (const float*)d_in[17];
  const float* bn2_m = (const float*)d_in[18];
  const float* bn2_v = (const float*)d_in[19];
  float* out = (float*)d_out;

  dim3 grid(32 * 64);   // one block per (batch, row)
  dim3 block(512);      // 16 wave32
  hipLaunchKernelGGL(dual_modal_expert_kernel, grid, block, 0, stream,
      x, wts, idx, dw_w0, pw_w0, bn0_s, bn0_b, bn0_m, bn0_v,
      dw_w1, pw_w1, bn1_s, bn1_b, bn1_m, bn1_v,
      w2, bn2_s, bn2_b, bn2_m, bn2_v, out);
}
